// VAE_81458349736055
// MI455X (gfx1250) — matrix operand order repaired and checked by reading, not compiled, not run on hardware
//
#include <hip/hip_runtime.h>

#define B_  1024
#define T_  15
#define E_  128
#define H_  1024
#define L_  128
#define V_  21
#define H4  4096
#define AH  1280           // E + H + L
#define TB  (T_ * B_)
#define BETA_ 0.1f

typedef _Float16 half8  __attribute__((ext_vector_type(8)));
typedef _Float16 half16 __attribute__((ext_vector_type(16)));
typedef float    f32x8  __attribute__((ext_vector_type(8)));

__device__ __forceinline__ half16 cat16(half8 lo, half8 hi) {
  return __builtin_shufflevector(lo, hi, 0,1,2,3,4,5,6,7,8,9,10,11,12,13,14,15);
}
__device__ __forceinline__ float sigm(float x) { return 1.0f / (1.0f + expf(-x)); }

// ---------------------------------------------------------------------------
// K-accumulation core: acc[NT] += A16[16 x K] @ Bt16[K x 16*NT] for one wave.
// A fragment (16x32 f16): lane m = lane&15, grp = lane>>4;
//   halves 0..7  = K[grp*8 .. grp*8+7], halves 8..15 = K[16+grp*8 ..].
// B fragment (32x16 f16): lane n = lane&15, grp = lane>>4; same K split as A.
// All NT B-fragments are loaded into distinct registers before the WMMAs so
// the loads clause-batch and waits stagger instead of fully draining.
// ---------------------------------------------------------------------------
template <int NT>
__device__ __forceinline__ void mm_accum(const _Float16* __restrict__ A, int lda,
                                         const _Float16* __restrict__ Bt, int ldb,
                                         int K, int mrow, int n0, int lane,
                                         f32x8 acc[NT]) {
  const int grp = lane >> 4;
  const _Float16* Ar = A + (size_t)mrow * lda + grp * 8;
  const _Float16* Br = Bt + (size_t)(grp * 8) * ldb + n0 + (lane & 15);
  const size_t bstep = (size_t)32 * ldb;
  for (int k0 = 0; k0 < K; k0 += 32) {
    half16 a = cat16(*(const half8*)(Ar), *(const half8*)(Ar + 16));
    half16 b[NT];
#pragma unroll
    for (int nt = 0; nt < NT; ++nt)
#pragma unroll
      for (int s = 0; s < 16; ++s)
        b[nt][s] = Br[(size_t)(s < 8 ? s : s + 8) * ldb + 16 * nt];
#pragma unroll
    for (int nt = 0; nt < NT; ++nt)
      acc[nt] = __builtin_amdgcn_wmma_f32_16x16x32_f16(false, a, false, b[nt],
                                                       (short)0, acc[nt], false, false);
    Ar += 32;
    Br += bstep;
  }
}

// ---------------------------------------------------------------------------
// Fused dual-operand WMMA GEMM:
//   C[M x N] = A1 @ B1t (+ A2 @ B2t) + bias   [relu] [f32 and/or f16 stores]
// grid = (N/(16*NT), M/64), block = 128 threads (4 waves), wave = 16 x 16*NT.
// C/D (16x16 f32): lane n = lane&15, vgpr v -> m = v + 8*(lane>>4).
// ---------------------------------------------------------------------------
template <int NT>
__global__ __launch_bounds__(128) void gemm_wmma(
    const _Float16* __restrict__ A1, int lda1,
    const _Float16* __restrict__ B1, int ldb1, int K1,
    const _Float16* __restrict__ A2, int lda2,
    const _Float16* __restrict__ B2, int ldb2, int K2,
    float* __restrict__ C, int ldc,
    _Float16* __restrict__ C16, int ldc16,
    const float* __restrict__ bias, int relu)
{
  const int lane = threadIdx.x & 31;
  const int wave = threadIdx.x >> 5;
  const int n0 = blockIdx.x * (16 * NT);
  const int m0 = blockIdx.y * 64 + wave * 16;
  const int nl = lane & 15;
  const int grp = lane >> 4;

  f32x8 acc[NT];
  if (bias) {
#pragma unroll
    for (int nt = 0; nt < NT; ++nt) {
      float bv = bias[n0 + 16 * nt + nl];
#pragma unroll
      for (int v = 0; v < 8; ++v) acc[nt][v] = bv;
    }
  } else {
#pragma unroll
    for (int nt = 0; nt < NT; ++nt)
#pragma unroll
      for (int v = 0; v < 8; ++v) acc[nt][v] = 0.f;
  }

  mm_accum<NT>(A1, lda1, B1, ldb1, K1, m0 + nl, n0, lane, acc);
  if (A2) mm_accum<NT>(A2, lda2, B2, ldb2, K2, m0 + nl, n0, lane, acc);

#pragma unroll
  for (int nt = 0; nt < NT; ++nt)
#pragma unroll
    for (int v = 0; v < 8; ++v) {
      int m = m0 + v + 8 * grp;
      float x = acc[nt][v];
      if (relu) x = fmaxf(x, 0.f);
      if (C)   C[(size_t)m * ldc + n0 + 16 * nt + nl] = x;
      if (C16) C16[(size_t)m * ldc16 + n0 + 16 * nt + nl] = (_Float16)x;
    }
}

// ------------------------------- elementwise --------------------------------
__global__ void k_fill_f32(float* p, long n) {
  for (long i = (long)blockIdx.x * blockDim.x + threadIdx.x; i < n;
       i += (long)gridDim.x * blockDim.x) p[i] = 0.f;
}
__global__ void k_fill_f16(_Float16* p, long n) {
  for (long i = (long)blockIdx.x * blockDim.x + threadIdx.x; i < n;
       i += (long)gridDim.x * blockDim.x) p[i] = (_Float16)0.f;
}
__global__ void k_cvt16(const float* __restrict__ s, _Float16* __restrict__ d, long n) {
  long i = (long)blockIdx.x * blockDim.x + threadIdx.x;
  if (i < n) d[i] = (_Float16)s[i];
}
// out[k*ld_out + col_off + n] = (f16) W[n*K + k]   (W is N x K row-major)
__global__ void k_transpose(const float* __restrict__ W, _Float16* __restrict__ out,
                            int N, int K, int ld_out, int col_off) {
  long i = (long)blockIdx.x * blockDim.x + threadIdx.x;
  if (i >= (long)N * K) return;
  int n = (int)(i / K), k = (int)(i % K);
  out[(size_t)k * ld_out + col_off + n] = (_Float16)W[i];
}
__global__ void k_xrev(const float* __restrict__ x, const int* __restrict__ len,
                       _Float16* __restrict__ out) {
  long i = (long)blockIdx.x * blockDim.x + threadIdx.x;
  if (i >= (long)B_ * T_ * E_) return;
  int b = (int)(i / (T_ * E_));
  int r = (int)(i % (T_ * E_));
  int t = r / E_, e = r % E_;
  int ri = len[b] - 1 - t;
  ri = ri < 0 ? 0 : (ri > T_ - 1 ? T_ - 1 : ri);
  out[i] = (_Float16)x[(size_t)b * T_ * E_ + (size_t)ri * E_ + e];
}
__global__ void k_xdec(const float* __restrict__ emb, const int* __restrict__ tokens,
                       _Float16* __restrict__ out) {
  long i = (long)blockIdx.x * blockDim.x + threadIdx.x;
  if (i >= (long)B_ * T_ * E_) return;
  int b = (int)(i / (T_ * E_));
  int r = (int)(i % (T_ * E_));
  int t = r / E_, e = r % E_;
  if (t == 0) { out[i] = (_Float16)0.f; return; }
  int tok = tokens[b * T_ + (t - 1)];
  out[i] = (_Float16)emb[(size_t)tok * E_ + e];
}
// encoder gate order (jnp.split): [i | f | g | o]
__global__ void k_enc_update(const float* __restrict__ g, float* __restrict__ h,
                             float* __restrict__ c, _Float16* __restrict__ h16,
                             const int* __restrict__ len, int t) {
  long i = (long)blockIdx.x * blockDim.x + threadIdx.x;
  if (i >= (long)B_ * H_) return;
  int b = (int)(i / H_), j = (int)(i % H_);
  const float* gr = g + (size_t)b * H4;
  float ig = sigm(gr[j]);
  float fg = sigm(gr[H_ + j]);
  float gg = tanhf(gr[2 * H_ + j]);
  float og = sigm(gr[3 * H_ + j]);
  float co = c[i], ho = h[i];
  float nc = fg * co + ig * gg;
  float nh = og * tanhf(nc);
  bool m = t < len[b];
  float hn = m ? nh : ho, cn = m ? nc : co;
  h[i] = hn; c[i] = cn; h16[i] = (_Float16)hn;
}
// decoder gate order in Wcat columns: [i | o | f | g]; also writes ahs h-part (OLD h)
__global__ void k_dec_update(const float* __restrict__ g, float* __restrict__ h,
                             float* __restrict__ c, _Float16* __restrict__ h16,
                             _Float16* __restrict__ ahs, int t) {
  long i = (long)blockIdx.x * blockDim.x + threadIdx.x;
  if (i >= (long)B_ * H_) return;
  int b = (int)(i / H_), j = (int)(i % H_);
  const float* gr = g + (size_t)b * H4;
  float ig = sigm(gr[j]);
  float og = sigm(gr[H_ + j]);
  float fg = sigm(gr[2 * H_ + j]);
  float gg = tanhf(gr[3 * H_ + j]);
  float nc = fg * c[i] + ig * gg;
  float nh = og * tanhf(nc);
  ahs[((size_t)t * B_ + b) * AH + E_ + j] = h16[i];   // pre-update h
  h[i] = nh; c[i] = nc; h16[i] = (_Float16)nh;
}
// ahs x-part and z-part for all timesteps
__global__ void k_ahs_xz(const _Float16* __restrict__ xdec16,
                         const _Float16* __restrict__ z16,
                         _Float16* __restrict__ ahs) {
  long i = (long)blockIdx.x * blockDim.x + threadIdx.x;
  long per = (long)(E_ + L_);
  if (i >= (long)TB * per) return;
  long row = i / per;
  int p = (int)(i % per);
  int t = (int)(row / B_), b = (int)(row % B_);
  _Float16* dst = ahs + row * AH;
  if (p < E_) dst[p] = xdec16[(size_t)b * T_ * E_ + (size_t)t * E_ + p];
  else        dst[E_ + H_ + (p - E_)] = z16[(size_t)b * L_ + (p - E_)];
}
__global__ void k_hidden_cat(const float* __restrict__ hf, const float* __restrict__ hb,
                             _Float16* __restrict__ out) {
  long i = (long)blockIdx.x * blockDim.x + threadIdx.x;
  if (i >= (long)B_ * 2 * H_) return;
  int b = (int)(i / (2 * H_)), j = (int)(i % (2 * H_));
  float v = (j < H_) ? hf[(size_t)b * H_ + j] : hb[(size_t)b * H_ + (j - H_)];
  out[i] = (_Float16)v;
}
__global__ void k_z(const float* __restrict__ eps, const float* __restrict__ mean,
                    const float* __restrict__ logv, _Float16* __restrict__ z16) {
  long i = (long)blockIdx.x * blockDim.x + threadIdx.x;
  if (i >= (long)B_ * L_) return;
  float z = eps[i] * expf(0.5f * logv[i]) + mean[i];
  z16[i] = (_Float16)z;
}
__global__ void k_bcat(const float* bi, const float* bo, const float* bf,
                       const float* bg, float* out) {
  int j = blockIdx.x * blockDim.x + threadIdx.x;
  if (j >= H4) return;
  float v;
  if (j < H_)            v = bi[j];
  else if (j < 2 * H_)   v = bo[j - H_];
  else if (j < 3 * H_)   v = bf[j - 2 * H_];
  else                   v = bg[j - 3 * H_];
  out[j] = v;
}
__global__ void k_b2pad(const float* b2, float* out) {
  int j = threadIdx.x;
  if (j < 32) out[j] = (j < V_) ? b2[j] : 0.f;
}
// deterministic single-block final reduction -> loss, amino, kl, acc
__global__ void k_finalize(const float* __restrict__ scores, const int* __restrict__ tokens,
                           const int* __restrict__ lengths, const float* __restrict__ mean,
                           const float* __restrict__ logv, float* __restrict__ out) {
  __shared__ float sh[4][256];
  int tid = threadIdx.x;
  float kl_s = 0.f, ce_s = 0.f, cor_s = 0.f, mk_s = 0.f;
  for (int i = tid; i < B_ * L_; i += 256) {
    float m = mean[i], lv = logv[i];
    kl_s += 1.0f + lv - m * m - expf(lv);
  }
  for (int r = tid; r < TB; r += 256) {
    int t = r / B_, b = r - t * B_;
    const float* s = scores + (size_t)r * 32;
    float mx = s[0]; int am = 0;
    for (int v = 1; v < V_; ++v) { float sv = s[v]; if (sv > mx) { mx = sv; am = v; } }
    float se = 0.f;
    for (int v = 0; v < V_; ++v) se += expf(s[v] - mx);
    int len = lengths[b];
    int tgt = (t < len) ? tokens[b * T_ + t] : 0;
    float mkv = (t <= len) ? 1.f : 0.f;
    float ce = -(s[tgt] - mx - logf(se));
    ce_s += ce * mkv;
    cor_s += (am == tgt) ? mkv : 0.f;
    mk_s += mkv;
  }
  sh[0][tid] = kl_s; sh[1][tid] = ce_s; sh[2][tid] = cor_s; sh[3][tid] = mk_s;
  __syncthreads();
  for (int st = 128; st > 0; st >>= 1) {
    if (tid < st)
      for (int q = 0; q < 4; ++q) sh[q][tid] += sh[q][tid + st];
    __syncthreads();
  }
  if (tid == 0) {
    float kl = -0.5f * sh[0][0] / (float)B_;
    float amino = sh[1][0] / (float)B_;
    float acc = sh[2][0] / sh[3][0];
    out[0] = amino + BETA_ * kl;
    out[1] = amino;
    out[2] = kl;
    out[3] = acc;
  }
}

// ---------------------------------------------------------------------------
static inline void gemm2(hipStream_t s,
                         const _Float16* A1, int lda1, const _Float16* B1, int ldb1, int K1,
                         const _Float16* A2, int lda2, const _Float16* B2, int ldb2, int K2,
                         float* C, int ldc, _Float16* C16, int ldc16,
                         const float* bias, int M, int N, int relu) {
  if ((N & 63) == 0) {
    dim3 g(N / 64, M / 64);
    gemm_wmma<4><<<g, 128, 0, s>>>(A1, lda1, B1, ldb1, K1, A2, lda2, B2, ldb2, K2,
                                   C, ldc, C16, ldc16, bias, relu);
  } else {
    dim3 g(N / 32, M / 64);
    gemm_wmma<2><<<g, 128, 0, s>>>(A1, lda1, B1, ldb1, K1, A2, lda2, B2, ldb2, K2,
                                   C, ldc, C16, ldc16, bias, relu);
  }
}
static inline int blks(long n) { return (int)((n + 255) / 256); }

extern "C" void kernel_launch(void* const* d_in, const int* in_sizes, int n_in,
                              void* d_out, int out_size, void* d_ws, size_t ws_size,
                              hipStream_t stream) {
  const float* x_enc   = (const float*)d_in[0];
  const int*   tokens  = (const int*)d_in[1];
  const int*   lengths = (const int*)d_in[2];
  const float* eps     = (const float*)d_in[3];
  const float* emb     = (const float*)d_in[4];
  const float* Wih_f   = (const float*)d_in[5];
  const float* Whh_f   = (const float*)d_in[6];
  const float* b_f     = (const float*)d_in[7];
  const float* Wih_b   = (const float*)d_in[8];
  const float* Whh_b   = (const float*)d_in[9];
  const float* b_b     = (const float*)d_in[10];
  const float* Wm      = (const float*)d_in[11];
  const float* bm      = (const float*)d_in[12];
  const float* Wl      = (const float*)d_in[13];
  const float* bl      = (const float*)d_in[14];
  const float* Wz      = (const float*)d_in[15];
  const float* bz      = (const float*)d_in[16];
  const float* Wi      = (const float*)d_in[17];
  const float* bi      = (const float*)d_in[18];
  const float* Wo      = (const float*)d_in[19];
  const float* bo      = (const float*)d_in[20];
  const float* Wf      = (const float*)d_in[21];
  const float* bf      = (const float*)d_in[22];
  const float* Wg      = (const float*)d_in[23];
  const float* bg      = (const float*)d_in[24];
  const float* W1      = (const float*)d_in[25];
  const float* b1      = (const float*)d_in[26];
  const float* W2      = (const float*)d_in[27];
  const float* b2      = (const float*)d_in[28];
  float* out = (float*)d_out;

  char* base = (char*)d_ws;
  size_t cur = 0;
  auto alloc = [&](size_t bytes) -> char* {
    char* p = base + cur;
    cur += (bytes + 255) & ~(size_t)255;
    return p;
  };
  const size_t HS = sizeof(_Float16), FS = sizeof(float);

  _Float16* xenc16 = (_Float16*)alloc(HS * B_ * T_ * E_);
  _Float16* xrev16 = (_Float16*)alloc(HS * B_ * T_ * E_);
  _Float16* xdec16 = (_Float16*)alloc(HS * B_ * T_ * E_);
  _Float16* WihfT  = (_Float16*)alloc(HS * E_ * H4);
  _Float16* WhhfT  = (_Float16*)alloc(HS * H_ * H4);
  _Float16* WihbT  = (_Float16*)alloc(HS * E_ * H4);
  _Float16* WhhbT  = (_Float16*)alloc(HS * H_ * H4);
  _Float16* WmT    = (_Float16*)alloc(HS * 2 * H_ * L_);
  _Float16* WlT    = (_Float16*)alloc(HS * 2 * H_ * L_);
  _Float16* WzT    = (_Float16*)alloc(HS * L_ * H_);
  _Float16* WcatT  = (_Float16*)alloc(HS * (E_ + H_) * H4);
  _Float16* W1T    = (_Float16*)alloc(HS * AH * H_);
  _Float16* W2T    = (_Float16*)alloc(HS * H_ * 32);
  _Float16* h16    = (_Float16*)alloc(HS * B_ * H_);
  _Float16* hid16  = (_Float16*)alloc(HS * B_ * 2 * H_);
  _Float16* z16    = (_Float16*)alloc(HS * B_ * L_);
  _Float16* ahs16  = (_Float16*)alloc(HS * (size_t)TB * AH);
  _Float16* hid1   = (_Float16*)alloc(HS * (size_t)TB * H_);
  float* g4     = (float*)alloc(FS * (size_t)B_ * H4);
  float* hf     = (float*)alloc(FS * B_ * H_);
  float* hb     = (float*)alloc(FS * B_ * H_);
  float* cst    = (float*)alloc(FS * B_ * H_);
  float* mean   = (float*)alloc(FS * B_ * L_);
  float* logv   = (float*)alloc(FS * B_ * L_);
  float* scores = (float*)alloc(FS * (size_t)TB * 32);
  float* bcat   = (float*)alloc(FS * H4);
  float* b2pad  = (float*)alloc(FS * 32);

  // ------------------------- prep: conversions/transposes -------------------
  long nxe = (long)B_ * T_ * E_;
  k_cvt16<<<blks(nxe), 256, 0, stream>>>(x_enc, xenc16, nxe);
  k_xrev<<<blks(nxe), 256, 0, stream>>>(x_enc, lengths, xrev16);
  k_xdec<<<blks(nxe), 256, 0, stream>>>(emb, tokens, xdec16);
  k_transpose<<<blks((long)H4 * E_), 256, 0, stream>>>(Wih_f, WihfT, H4, E_, H4, 0);
  k_transpose<<<blks((long)H4 * H_), 256, 0, stream>>>(Whh_f, WhhfT, H4, H_, H4, 0);
  k_transpose<<<blks((long)H4 * E_), 256, 0, stream>>>(Wih_b, WihbT, H4, E_, H4, 0);
  k_transpose<<<blks((long)H4 * H_), 256, 0, stream>>>(Whh_b, WhhbT, H4, H_, H4, 0);
  k_transpose<<<blks((long)L_ * 2 * H_), 256, 0, stream>>>(Wm, WmT, L_, 2 * H_, L_, 0);
  k_transpose<<<blks((long)L_ * 2 * H_), 256, 0, stream>>>(Wl, WlT, L_, 2 * H_, L_, 0);
  k_transpose<<<blks((long)H_ * L_), 256, 0, stream>>>(Wz, WzT, H_, L_, H_, 0);
  k_transpose<<<blks((long)H_ * (E_ + H_)), 256, 0, stream>>>(Wi, WcatT, H_, E_ + H_, H4, 0);
  k_transpose<<<blks((long)H_ * (E_ + H_)), 256, 0, stream>>>(Wo, WcatT, H_, E_ + H_, H4, H_);
  k_transpose<<<blks((long)H_ * (E_ + H_)), 256, 0, stream>>>(Wf, WcatT, H_, E_ + H_, H4, 2 * H_);
  k_transpose<<<blks((long)H_ * (E_ + H_)), 256, 0, stream>>>(Wg, WcatT, H_, E_ + H_, H4, 3 * H_);
  k_transpose<<<blks((long)H_ * AH), 256, 0, stream>>>(W1, W1T, H_, AH, H_, 0);
  k_fill_f16<<<blks((long)H_ * 32), 256, 0, stream>>>(W2T, (long)H_ * 32);
  k_transpose<<<blks((long)V_ * H_), 256, 0, stream>>>(W2, W2T, V_, H_, 32, 0);
  k_bcat<<<blks(H4), 256, 0, stream>>>(bi, bo, bf, bg, bcat);
  k_b2pad<<<1, 32, 0, stream>>>(b2, b2pad);

  long nbh = (long)B_ * H_;
  // ------------------------------ encoder fwd -------------------------------
  k_fill_f32<<<blks(nbh), 256, 0, stream>>>(hf, nbh);
  k_fill_f32<<<blks(nbh), 256, 0, stream>>>(cst, nbh);
  k_fill_f16<<<blks(nbh), 256, 0, stream>>>(h16, nbh);
  for (int t = 0; t < T_; ++t) {
    gemm2(stream, xenc16 + (size_t)t * E_, T_ * E_, WihfT, H4, E_,
          h16, H_, WhhfT, H4, H_,
          g4, H4, nullptr, 0, b_f, B_, H4, 0);
    k_enc_update<<<blks(nbh), 256, 0, stream>>>(g4, hf, cst, h16, lengths, t);
  }
  // ------------------------------ encoder bwd -------------------------------
  k_fill_f32<<<blks(nbh), 256, 0, stream>>>(hb, nbh);
  k_fill_f32<<<blks(nbh), 256, 0, stream>>>(cst, nbh);
  k_fill_f16<<<blks(nbh), 256, 0, stream>>>(h16, nbh);
  for (int t = 0; t < T_; ++t) {
    gemm2(stream, xrev16 + (size_t)t * E_, T_ * E_, WihbT, H4, E_,
          h16, H_, WhhbT, H4, H_,
          g4, H4, nullptr, 0, b_b, B_, H4, 0);
    k_enc_update<<<blks(nbh), 256, 0, stream>>>(g4, hb, cst, h16, lengths, t);
  }
  // ------------------------------- latent -----------------------------------
  k_hidden_cat<<<blks((long)B_ * 2 * H_), 256, 0, stream>>>(hf, hb, hid16);
  gemm2(stream, hid16, 2 * H_, WmT, L_, 2 * H_,
        nullptr, 0, nullptr, 0, 0, mean, L_, nullptr, 0, bm, B_, L_, 0);
  gemm2(stream, hid16, 2 * H_, WlT, L_, 2 * H_,
        nullptr, 0, nullptr, 0, 0, logv, L_, nullptr, 0, bl, B_, L_, 0);
  k_z<<<blks((long)B_ * L_), 256, 0, stream>>>(eps, mean, logv, z16);
  // ------------------------------- decoder ----------------------------------
  gemm2(stream, z16, L_, WzT, H_, L_,
        nullptr, 0, nullptr, 0, 0, hf, H_, h16, H_, bz, B_, H_, 0);   // h0
  k_fill_f32<<<blks(nbh), 256, 0, stream>>>(cst, nbh);                // c0
  k_ahs_xz<<<blks((long)TB * (E_ + L_)), 256, 0, stream>>>(xdec16, z16, ahs16);
  for (int t = 0; t < T_; ++t) {
    gemm2(stream, xdec16 + (size_t)t * E_, T_ * E_, WcatT, H4, E_,
          h16, H_, WcatT + (size_t)E_ * H4, H4, H_,
          g4, H4, nullptr, 0, bcat, B_, H4, 0);
    k_dec_update<<<blks(nbh), 256, 0, stream>>>(g4, hf, cst, h16, ahs16, t);
  }
  // ------------------------------ output MLP --------------------------------
  gemm2(stream, ahs16, AH, W1T, H_, AH,
        nullptr, 0, nullptr, 0, 0, nullptr, 0, hid1, H_, b1, TB, H_, 1);
  gemm2(stream, hid1, H_, W2T, 32, H_,
        nullptr, 0, nullptr, 0, 0, scores, 32, nullptr, 0, b2pad, TB, 32, 0);
  // ------------------------------ reductions --------------------------------
  k_finalize<<<1, 256, 0, stream>>>(scores, tokens, lengths, mean, logv, out);
}
